// AdaptiveMMLDotProductGroundedCoreferencer_27075473834194
// MI455X (gfx1250) — compile-verified, matrix operand order checked by reading, not run on hardware
//
#include <hip/hip_runtime.h>
#include <hip/hip_bf16.h>

// ---------------------------------------------------------------------------
// CDNA5 (gfx1250) implementation. wave32. Matrix math via
// v_wmma_f32_16x16x32_bf16, fp32 accumulate. All WMMA operands are
// pre-packed bf16, K-padded to a multiple of 32, weights transposed so every
// lane fragment is two contiguous 16-byte runs -> b128 loads only.
// pair_scores uses 2-way M blocking so each B fragment feeds two WMMAs
// (halves L2 weight traffic on the dominant ~112 GFLOP stage).
// ---------------------------------------------------------------------------

typedef __attribute__((ext_vector_type(16))) __bf16 v16bf;
typedef __attribute__((ext_vector_type(8)))  __bf16 bf16x8;
typedef __attribute__((ext_vector_type(8)))  float  v8f;

#define NEGBIG (-1e10f)
#define KPAD   2368      // SD=2324 padded to multiple of 32 (and of 2*KCHUNK)
#define KCHUNK 1184      // K staged into LDS in two chunks (37 wmma steps each)

static __device__ inline __bf16 f2bf(float f) {
    unsigned u = __builtin_bit_cast(unsigned, f);
    unsigned short h = (unsigned short)((u + 0x7FFFu + ((u >> 16) & 1u)) >> 16);
    return __builtin_bit_cast(__bf16, h);
}
static __device__ inline float bf2f(__bf16 b) {
    unsigned short h = __builtin_bit_cast(unsigned short, b);
    unsigned u = ((unsigned)h) << 16;
    return __builtin_bit_cast(float, u);
}

// Fragment load: per ISA 7.12.2 a lane of a 16x32 bf16 A/B fragment holds
// k = k0+hi*8 .. +7  and  k = k0+16+hi*8 .. +7  (hi = lane>>4).
// With contiguous bf16 rows these are two aligned 16-byte loads.
static __device__ inline v16bf load_frag(const __bf16* __restrict__ row, int k0, int hi) {
    bf16x8 lo = *(const bf16x8*)(row + k0 + hi * 8);
    bf16x8 hi8 = *(const bf16x8*)(row + k0 + 16 + hi * 8);
    return __builtin_shufflevector(lo, hi8, 0, 1, 2, 3, 4, 5, 6, 7,
                                   8, 9, 10, 11, 12, 13, 14, 15);
}
static __device__ inline v16bf lds_frag(const __bf16* row, int k0, int hi) {
    bf16x8 lo = *(const bf16x8*)(row + k0 + hi * 8);
    bf16x8 hi8 = *(const bf16x8*)(row + k0 + 16 + hi * 8);
    return __builtin_shufflevector(lo, hi8, 0, 1, 2, 3, 4, 5, 6, 7,
                                   8, 9, 10, 11, 12, 13, 14, 15);
}

// ---------------------------------------------------------------------------
// Packing kernels (one-shot, bandwidth-trivial).
// ---------------------------------------------------------------------------

// rows x K fp32 -> rows x Kpad bf16 (zero pad K..Kpad)
__global__ void pack_rows(const float* __restrict__ in, __bf16* __restrict__ out,
                          int K, int Kpad, long rows)
{
    long i = (long)blockIdx.x * blockDim.x + threadIdx.x;
    if (i >= rows * Kpad) return;
    long r = i / Kpad;
    int k = (int)(i % Kpad);
    out[i] = (k < K) ? f2bf(in[r * K + k]) : f2bf(0.0f);
}

// K x N fp32 weight -> N x Kpad bf16 (transposed, zero pad)
__global__ void pack_wt(const float* __restrict__ B, __bf16* __restrict__ out,
                        int K, int N, int Kpad)
{
    long i = (long)blockIdx.x * blockDim.x + threadIdx.x;
    if (i >= (long)N * Kpad) return;
    int n = (int)(i / Kpad);
    int k = (int)(i % Kpad);
    out[i] = (k < K) ? f2bf(B[(long)k * N + n]) : f2bf(0.0f);
}

// img (8,36,1024) fp32 -> (8,48,1024) bf16, rows 36..47 zeroed
__global__ void pack_img(const float* __restrict__ img, __bf16* __restrict__ out)
{
    long i = (long)blockIdx.x * blockDim.x + threadIdx.x;
    const long total = 8L * 48 * 1024;
    if (i >= total) return;
    int k = (int)(i & 1023);
    long r = i >> 10;          // 0..383
    int n = (int)(r % 48);
    long v = r / 48;
    out[i] = (n < 36) ? f2bf(img[(v * 36 + n) * 1024 + k]) : f2bf(0.0f);
}

// ---------------------------------------------------------------------------
// Packed WMMA GEMM: C[M,N] = act(A@B + bias).
//   A  : M x Kpad bf16 row-major (K-padded)
//   Bt : N x Kpad bf16 row-major (weights pre-transposed)
// One wave per 16x16 C tile; branch-free inner loop, b128 loads only.
// Grids are sized so m0+15 < M and n0+15 < N always.
// ---------------------------------------------------------------------------
__global__ __launch_bounds__(32)
void gemm_packed(const __bf16* __restrict__ A, int ldak,
                 const __bf16* __restrict__ Bt, int ldbk,
                 float* __restrict__ C, int ldc,
                 const float* __restrict__ bias,
                 int Kpad, int relu)
{
    int lane = threadIdx.x & 31;
    int hi = (lane >> 4) & 1;
    int mn = lane & 15;
    int m0 = blockIdx.x * 16, n0 = blockIdx.y * 16;

    const __bf16* arow = A + (long)(m0 + mn) * ldak;
    const __bf16* brow = Bt + (long)(n0 + mn) * ldbk;

    v8f acc = {};
    for (int k0 = 0; k0 < Kpad; k0 += 32) {
        __builtin_prefetch(brow + k0 + 128, 0, 1);
        v16bf a = load_frag(arow, k0, hi);
        v16bf b = load_frag(brow, k0, hi);
        acc = __builtin_amdgcn_wmma_f32_16x16x32_bf16(false, a, false, b,
                                                      (short)0, acc, false, false);
    }
    int n = n0 + mn;
    float bv = bias ? bias[n] : 0.0f;
#pragma unroll
    for (int r = 0; r < 8; ++r) {
        int m = m0 + r + hi * 8;
        float v = acc[r] + bv;
        if (relu) v = fmaxf(v, 0.0f);
        C[(long)m * ldc + n] = v;
    }
}

// ---------------------------------------------------------------------------
// Span embedder tail: per-span token scores, masked softmax over tokens,
// weighted sum + width embedding, concat. One block (10 waves) per span.
// ---------------------------------------------------------------------------
__global__ __launch_bounds__(320)
void span_finish(const float* __restrict__ h_full,   // (n*MS*W, 1024)
                 const float* __restrict__ w2,       // (1024)
                 const float* __restrict__ b2,       // (1)
                 const int*   __restrict__ width,    // (n*MS)
                 const float* __restrict__ se,       // (n*MS, 1536)
                 const float* __restrict__ cont,     // (n*MS, W, 768)
                 const float* __restrict__ wemb,     // (5, 20)
                 float* __restrict__ spans)          // (n*MS, 2324)
{
    const int W = 10, H = 1024, BH = 768, ED = 20, SD = 2324;
    int span = blockIdx.x;
    int tid = threadIdx.x;
    int wave = tid >> 5, lane = tid & 31;

    __shared__ float s_sc[W];
    __shared__ float s_attn[W];

    if (wave < W) {
        const float* hrow = h_full + ((long)span * W + wave) * H;
        float acc = 0.0f;
        for (int k = lane; k < H; k += 32) acc += hrow[k] * w2[k];
        for (int off = 16; off > 0; off >>= 1) acc += __shfl_down(acc, off);
        if (lane == 0) s_sc[wave] = acc + b2[0];
    }
    __syncthreads();
    if (tid == 0) {
        int wd = width[span];
        float mx = -3.4e38f;
        for (int t = 0; t < W; ++t) {
            float s = (t < wd) ? s_sc[t] : NEGBIG;
            s_sc[t] = s;
            mx = fmaxf(mx, s);
        }
        float den = 0.0f;
        for (int t = 0; t < W; ++t) { float e = __expf(s_sc[t] - mx); s_attn[t] = e; den += e; }
        for (int t = 0; t < W; ++t) s_attn[t] /= den;
    }
    __syncthreads();

    float* out = spans + (long)span * SD;
    const float* serow = se + (long)span * 1536;
    for (int d = tid; d < 1536; d += blockDim.x) out[d] = serow[d];
    const float* crow = cont + (long)span * W * BH;
    for (int d = tid; d < BH; d += blockDim.x) {
        float acc = 0.0f;
#pragma unroll
        for (int t = 0; t < W; ++t) acc += s_attn[t] * crow[t * BH + d];
        out[1536 + d] = acc;
    }
    int wd = width[span];
    int wi = wd < 0 ? 0 : (wd > 4 ? 4 : wd);
    for (int d = tid; d < ED; d += blockDim.x) out[1536 + BH + d] = wemb[wi * ED + d];
}

// ---------------------------------------------------------------------------
// Grounding: att = doc[s] @ img[v]^T via WMMA (packed bf16, padded img rows),
// then the masked double-softmax score. One block (8 waves) per (s,v).
// ---------------------------------------------------------------------------
__global__ __launch_bounds__(256)
void grounding(const __bf16* __restrict__ doc_bf,  // (8,64,1024)
               const __bf16* __restrict__ imgp,    // (8,48,1024) rows 36+ zero
               const float* __restrict__ tm,       // (8,64)
               const float* __restrict__ im,       // (8,36)
               float* __restrict__ S_g)            // (8,8)
{
    const int F = 64, R = 36, D = 1024;
    int s = blockIdx.x >> 3, v = blockIdx.x & 7;
    int tid = threadIdx.x, wave = tid >> 5, lane = tid & 31;
    int hi = (lane >> 4) & 1, mn = lane & 15;

    __shared__ float s_att[F][R];
    __shared__ float s_acc[2];

    // Phase 1: 4 M-tiles x 3 N-tiles (N padded to 48)
    for (int t = wave; t < 12; t += 8) {
        int m0 = (t / 3) * 16, n0 = (t % 3) * 16;
        const __bf16* arow = doc_bf + ((long)s * F + m0 + mn) * D;
        const __bf16* brow = imgp + ((long)v * 48 + n0 + mn) * D;
        v8f acc = {};
        for (int k0 = 0; k0 < D; k0 += 32) {
            v16bf a = load_frag(arow, k0, hi);
            v16bf b = load_frag(brow, k0, hi);
            acc = __builtin_amdgcn_wmma_f32_16x16x32_bf16(false, a, false, b,
                                                          (short)0, acc, false, false);
        }
        int n = n0 + mn;
        if (n < R) {
#pragma unroll
            for (int r = 0; r < 8; ++r) s_att[m0 + r + hi * 8][n] = acc[r];
        }
    }
    if (tid < 2) s_acc[tid] = 0.0f;
    __syncthreads();

    // mask + NEG substitution
    for (int idx = tid; idx < F * R; idx += 256) {
        int i = idx / R, j = idx % R;
        float a = s_att[i][j] * tm[s * F + i] * im[v * R + j];
        s_att[i][j] = (a != 0.0f) ? a : NEGBIG;
    }
    __syncthreads();

    // softmax over image axis (j), logits weighted by image mask
    if (tid < F) {
        int i = tid;
        float mx = -3.4e38f;
        for (int j = 0; j < R; ++j) mx = fmaxf(mx, s_att[i][j] * im[v * R + j]);
        float den = 0.0f;
        for (int j = 0; j < R; ++j) den += __expf(s_att[i][j] * im[v * R + j] - mx);
        float s1 = 0.0f;
        for (int j = 0; j < R; ++j) {
            float aw = __expf(s_att[i][j] * im[v * R + j] - mx) / den;
            aw *= tm[s * F + i] * im[v * R + j];
            s1 += aw * s_att[i][j];
        }
        atomicAdd(&s_acc[0], s1);
    }
    // softmax over frame axis (i)
    if (tid >= 64 && tid < 64 + R) {
        int j = tid - 64;
        float mx = -3.4e38f;
        for (int i = 0; i < F; ++i) mx = fmaxf(mx, s_att[i][j]);
        float den = 0.0f;
        for (int i = 0; i < F; ++i) den += __expf(s_att[i][j] - mx);
        float s2 = 0.0f;
        for (int i = 0; i < F; ++i) {
            float aw = __expf(s_att[i][j] - mx) / den;
            aw *= tm[s * F + i] * im[v * R + j];
            s2 += aw * s_att[i][j];
        }
        atomicAdd(&s_acc[1], s2);
    }
    __syncthreads();
    if (tid == 0) S_g[s * 8 + v] = s_acc[0] + s_acc[1];
}

// ---------------------------------------------------------------------------
// Pairwise classifier (hot path, ~112 GFLOP).
// Block = (p-pair {2pb, 2pb+1}, q-tile of 16)  ->  32 pair rows per block.
// Each B fragment (w1c_t / w2_t, from L2) feeds TWO WMMAs (one per p) ->
// 2x arithmetic intensity on the L2 weight stream vs 1 M-tile per block.
//   stage  : s_prod[32][KCHUNK] = span_p (.) span_q  (bf16, 74 KB LDS),
//            K processed in 2 chunks; per wave 4 n-tiles x 2 p accumulators
//            (64 VGPRs) stay live across chunks; 2 passes cover all 64 n-tiles.
//   phase 1: h1 = relu(prod @ W1c + a[p]+b[q]+b1) -> s_h1[32][1024] bf16 (64 KB)
//   phase 2: score[m] = sum_n relu(h1@W2+b2)[m,n]*w3[n]  (shuffle + ds_add)
// Total LDS ~138 KB / workgroup.
// ---------------------------------------------------------------------------
__global__ __launch_bounds__(256)
void pair_scores(const __bf16* __restrict__ spans_bf, // (128, KPAD) padded
                 const float*  __restrict__ a_vec,    // (128, 1024)
                 const float*  __restrict__ b_vec,    // (128, 1024)
                 const __bf16* __restrict__ w1c_t,    // (1024, KPAD) transposed
                 const __bf16* __restrict__ w2_t,     // (1024, 1024) transposed
                 const float*  __restrict__ b1,
                 const float*  __restrict__ b2,
                 const float*  __restrict__ b3,
                 const float*  __restrict__ w3,       // (1024)
                 float* __restrict__ ts)              // (8,8,16,16)
{
    const int H = 1024;
    int p0 = blockIdx.x * 2;  // pair rows for p0 and p0+1
    int qb = blockIdx.y;      // 0..7; q = qb*16 + m
    int tid = threadIdx.x;
    int wave = tid >> 5, lane = tid & 31;
    int hi = (lane >> 4) & 1, mn = lane & 15;

    __shared__ __bf16 s_prod[32][KCHUNK];  // 74 KB: rows 0-15 = p0, 16-31 = p0+1
    __shared__ __bf16 s_h1[32][H];         // 64 KB
    __shared__ float  s_score[32];

    if (tid < 32) s_score[tid] = 0.0f;

    // ---------------- phase 1: two passes over N, two K chunks ----------------
    for (int pass = 0; pass < 2; ++pass) {
        v8f acc0[4] = {};   // p0 accumulators, 4 n-tiles
        v8f acc1[4] = {};   // p0+1 accumulators
        for (int kc = 0; kc < 2; ++kc) {
            __syncthreads();
            // stage product chunk into LDS
            const int chunks = KCHUNK / 8;   // 148 bf16x8 chunks per row
            for (int idx = tid; idx < 32 * chunks; idx += 256) {
                int r = idx / chunks;        // 0..31
                int c = idx % chunks;
                int pi = r >> 4;
                int q = qb * 16 + (r & 15);
                long ko = (long)kc * KCHUNK + c * 8;
                bf16x8 pv = *(const bf16x8*)(spans_bf + (long)(p0 + pi) * KPAD + ko);
                bf16x8 qv = *(const bf16x8*)(spans_bf + (long)q * KPAD + ko);
                bf16x8 rr;
#pragma unroll
                for (int e = 0; e < 8; ++e) rr[e] = f2bf(bf2f(pv[e]) * bf2f(qv[e]));
                *(bf16x8*)(&s_prod[r][c * 8]) = rr;
            }
            __syncthreads();
#pragma unroll
            for (int i = 0; i < 4; ++i) {
                int nt = wave + 8 * (pass * 4 + i);
                int n0 = nt * 16;
                const __bf16* brow = w1c_t + (long)(n0 + mn) * KPAD + kc * KCHUNK;
                for (int k0 = 0; k0 < KCHUNK; k0 += 32) {
                    __builtin_prefetch(brow + k0 + 128, 0, 1);
                    v16bf b  = load_frag(brow, k0, hi);
                    v16bf a0 = lds_frag(&s_prod[mn][0], k0, hi);
                    v16bf a1 = lds_frag(&s_prod[16 + mn][0], k0, hi);
                    acc0[i] = __builtin_amdgcn_wmma_f32_16x16x32_bf16(false, a0, false, b,
                                                                      (short)0, acc0[i], false, false);
                    acc1[i] = __builtin_amdgcn_wmma_f32_16x16x32_bf16(false, a1, false, b,
                                                                      (short)0, acc1[i], false, false);
                }
            }
        }
        // epilogue for this pass's 4 n-tiles (columns owned by this wave)
#pragma unroll
        for (int i = 0; i < 4; ++i) {
            int nt = wave + 8 * (pass * 4 + i);
            int n = nt * 16 + mn;
            float a0n = a_vec[(long)p0 * H + n];
            float a1n = a_vec[(long)(p0 + 1) * H + n];
            float b1n = b1[n];
#pragma unroll
            for (int r = 0; r < 8; ++r) {
                int m = r + hi * 8;
                int q = qb * 16 + m;
                float bqn = b_vec[(long)q * H + n];
                s_h1[m][n]      = f2bf(fmaxf(acc0[i][r] + a0n + bqn + b1n, 0.0f));
                s_h1[16 + m][n] = f2bf(fmaxf(acc1[i][r] + a1n + bqn + b1n, 0.0f));
            }
        }
    }
    __syncthreads();

    // ---------------- phase 2: K = H, fused w3 dot, shared B ----------------
    for (int nt = wave; nt < 64; nt += 8) {
        int n0 = nt * 16;
        const __bf16* brow = w2_t + (long)(n0 + mn) * H;
        v8f acc0 = {}, acc1 = {};
        for (int k0 = 0; k0 < H; k0 += 32) {
            __builtin_prefetch(brow + k0 + 128, 0, 1);
            v16bf b  = load_frag(brow, k0, hi);
            v16bf a0 = lds_frag(&s_h1[mn][0], k0, hi);
            v16bf a1 = lds_frag(&s_h1[16 + mn][0], k0, hi);
            acc0 = __builtin_amdgcn_wmma_f32_16x16x32_bf16(false, a0, false, b,
                                                           (short)0, acc0, false, false);
            acc1 = __builtin_amdgcn_wmma_f32_16x16x32_bf16(false, a1, false, b,
                                                           (short)0, acc1, false, false);
        }
        int n = n0 + mn;
        float w3n = w3[n], b2n = b2[n];
#pragma unroll
        for (int r = 0; r < 8; ++r) {
            float v0 = fmaxf(acc0[r] + b2n, 0.0f) * w3n;
            float v1 = fmaxf(acc1[r] + b2n, 0.0f) * w3n;
            for (int off = 8; off > 0; off >>= 1) {
                v0 += __shfl_xor(v0, off);
                v1 += __shfl_xor(v1, off);
            }
            if (mn == 0) {
                atomicAdd(&s_score[r + hi * 8], v0);
                atomicAdd(&s_score[16 + r + hi * 8], v1);
            }
        }
    }
    __syncthreads();
    if (tid < 32) {
        int pi = tid >> 4, j = tid & 15;
        int p = p0 + pi;
        int s = p >> 4, i = p & 15;
        ts[(((long)(s * 8 + qb)) * 16 + i) * 16 + j] = s_score[tid] + b3[0];
    }
}

// ---------------------------------------------------------------------------
// Adaptive S_c (tiny).
// ---------------------------------------------------------------------------
__global__ void adaptive_sc(const float* __restrict__ ts,
                            const float* __restrict__ m1,  // (8,16)
                            float* __restrict__ S_c)       // (8,8)
{
    int s = blockIdx.x >> 3, v = blockIdx.x & 7;
    if (threadIdx.x != 0) return;
    float cs = 0.0f, cv = 0.0f;
    for (int i = 0; i < 16; ++i) { cs += m1[s * 16 + i]; cv += m1[v * 16 + i]; }
    const float* t = ts + ((long)(s * 8 + v)) * 256;
    float max1 = -3.4e38f;
    for (int i = 0; i < 16; ++i) {
        if (m1[s * 16 + i] > 0.0f) {
            float rm = 0.0f;
            for (int j = 0; j < 16; ++j) rm += t[i * 16 + j] * m1[v * 16 + j];
            max1 = fmaxf(max1, rm / cv);
        }
    }
    float max2 = -3.4e38f;
    for (int j = 0; j < 16; ++j) {
        if (m1[v * 16 + j] > 0.0f) {
            float cm = 0.0f;
            for (int i = 0; i < 16; ++i) cm += t[i * 16 + j] * m1[s * 16 + i];
            max2 = fmaxf(max2, cm / cs);
        }
    }
    S_c[s * 8 + v] = 0.5f * (max1 + max2);
}

// ---------------------------------------------------------------------------
// Final 8x8 softmaxes + loss.
// ---------------------------------------------------------------------------
__global__ void final_loss(const float* __restrict__ S_g,
                           const float* __restrict__ S_c,
                           float* __restrict__ out)
{
    if (threadIdx.x != 0 || blockIdx.x != 0) return;
    const int N = 8;
    float mg[N * N], mgT[N * N], mc[N * N];
    auto softmax_rows = [&](const float* X, float* Y, bool transpose) {
        for (int r = 0; r < N; ++r) {
            float mx = -3.4e38f;
            for (int c = 0; c < N; ++c) {
                float x = transpose ? X[c * N + r] : X[r * N + c];
                mx = fmaxf(mx, x);
            }
            float den = 0.0f;
            for (int c = 0; c < N; ++c) {
                float x = transpose ? X[c * N + r] : X[r * N + c];
                float e = __expf(x - mx);
                Y[r * N + c] = e;
                den += e;
            }
            for (int c = 0; c < N; ++c) Y[r * N + c] /= den;
        }
    };
    softmax_rows(S_g, mg, false);
    softmax_rows(S_g, mgT, true);
    softmax_rows(S_c, mc, false);
    float loss = 0.0f;
    for (int s = 0; s < N; ++s) {
        float d1 = 0.0f, d2 = 0.0f;
        for (int v = 0; v < N; ++v) {
            d1 += mg[s * N + v] * mc[s * N + v];
            d2 += mgT[s * N + v] * mc[s * N + v];
        }
        loss -= __logf(d1) + __logf(d2);
    }
    out[0] = loss / (float)N;
}

// ---------------------------------------------------------------------------
// Host-side orchestration
// ---------------------------------------------------------------------------
extern "C" void kernel_launch(void* const* d_in, const int* in_sizes, int n_in,
                              void* d_out, int out_size, void* d_ws, size_t ws_size,
                              hipStream_t stream)
{
    (void)in_sizes; (void)n_in; (void)out_size; (void)ws_size;

    const float* doc   = (const float*)d_in[0];   // (8,64,1024)
    const float* img   = (const float*)d_in[1];   // (8,36,1024)
    const float* tm    = (const float*)d_in[2];   // (8,64)
    const float* im    = (const float*)d_in[3];   // (8,36)
    const float* se    = (const float*)d_in[4];   // (8,16,1536)
    const float* cont  = (const float*)d_in[5];   // (8,16,10,768)
    const int*   width = (const int*)  d_in[6];   // (8,16)
    const float* smask = (const float*)d_in[7];   // (8,16)
    const float* aw1   = (const float*)d_in[8];   // (768,1024)
    const float* ab1   = (const float*)d_in[9];   // (1024)
    const float* aw2   = (const float*)d_in[10];  // (1024,1)
    const float* ab2   = (const float*)d_in[11];  // (1)
    const float* wemb  = (const float*)d_in[12];  // (5,20)
    const float* pw_w1 = (const float*)d_in[13];  // (6972,1024)
    const float* pw_b1 = (const float*)d_in[14];  // (1024)
    const float* pw_w2 = (const float*)d_in[15];  // (1024,1024)
    const float* pw_b2 = (const float*)d_in[16];  // (1024)
    const float* pw_w3 = (const float*)d_in[17];  // (1024,1)
    const float* pw_b3 = (const float*)d_in[18];  // (1)

    const long SD = 2324, KP = KPAD, H = 1024;

    char* ws = (char*)d_ws;
    size_t off = 0;
    auto wsalloc = [&](size_t bytes) -> void* {
        void* pp = ws + off;
        off = (off + bytes + 255) & ~((size_t)255);
        return pp;
    };
    float*  h_full   = (float*) wsalloc(1280ull * 1024 * 4);
    float*  spans    = (float*) wsalloc(128ull * SD * 4);
    __bf16* spans_bf = (__bf16*)wsalloc(128ull * KP * 2);
    __bf16* cont_bf  = (__bf16*)wsalloc(1280ull * 768 * 2);
    __bf16* aw1_t    = (__bf16*)wsalloc(1024ull * 768 * 2);
    float*  a_vec    = (float*) wsalloc(128ull * H * 4);
    float*  b_vec    = (float*) wsalloc(128ull * H * 4);
    __bf16* w1a_t    = (__bf16*)wsalloc((size_t)H * KP * 2);
    __bf16* w1b_t    = (__bf16*)wsalloc((size_t)H * KP * 2);
    __bf16* w1c_t    = (__bf16*)wsalloc((size_t)H * KP * 2);
    __bf16* w2_t     = (__bf16*)wsalloc((size_t)H * H * 2);
    __bf16* doc_bf   = (__bf16*)wsalloc(512ull * 1024 * 2);
    __bf16* imgp_bf  = (__bf16*)wsalloc(8ull * 48 * 1024 * 2);
    float*  ts       = (float*) wsalloc(16384ull * 4);
    float*  S_g      = (float*) wsalloc(256);
    float*  S_c      = (float*) wsalloc(256);

    auto nb = [](long n) { return (unsigned)((n + 255) / 256); };

    // ---- packing ----
    pack_rows<<<nb(1280L * 768), 256, 0, stream>>>(cont, cont_bf, 768, 768, 1280);
    pack_wt  <<<nb(1024L * 768), 256, 0, stream>>>(aw1, aw1_t, 768, 1024, 768);
    pack_wt  <<<nb(H * KP), 256, 0, stream>>>(pw_w1,              w1a_t, (int)SD, (int)H, (int)KP);
    pack_wt  <<<nb(H * KP), 256, 0, stream>>>(pw_w1 + SD * H,     w1b_t, (int)SD, (int)H, (int)KP);
    pack_wt  <<<nb(H * KP), 256, 0, stream>>>(pw_w1 + 2 * SD * H, w1c_t, (int)SD, (int)H, (int)KP);
    pack_wt  <<<nb(H * H), 256, 0, stream>>>(pw_w2, w2_t, (int)H, (int)H, (int)H);
    pack_rows<<<nb(512L * 1024), 256, 0, stream>>>(doc, doc_bf, 1024, 1024, 512);
    pack_img <<<nb(8L * 48 * 1024), 256, 0, stream>>>(img, imgp_bf);

    // ---- span attention MLP: h_full = relu(cont @ aw1 + ab1) ----
    gemm_packed<<<dim3(80, 64), 32, 0, stream>>>(
        cont_bf, 768, aw1_t, 768, h_full, 1024, ab1, 768, 1);

    // ---- span scores/softmax/weighted-sum/concat -> spans (128 x 2324) ----
    span_finish<<<128, 320, 0, stream>>>(h_full, aw2, ab2, width, se, cont, wemb, spans);

    // ---- spans -> padded bf16 ----
    pack_rows<<<nb(128L * KP), 256, 0, stream>>>(spans, spans_bf, (int)SD, (int)KP, 128);

    // ---- pair projections a = spans @ W1[:SD], b = spans @ W1[SD:2SD] ----
    gemm_packed<<<dim3(8, 64), 32, 0, stream>>>(
        spans_bf, (int)KP, w1a_t, (int)KP, a_vec, (int)H, nullptr, (int)KP, 0);
    gemm_packed<<<dim3(8, 64), 32, 0, stream>>>(
        spans_bf, (int)KP, w1b_t, (int)KP, b_vec, (int)H, nullptr, (int)KP, 0);

    // ---- grounding S_g ----
    grounding<<<64, 256, 0, stream>>>(doc_bf, imgp_bf, tm, im, S_g);

    // ---- pairwise classifier -> ts (8,8,16,16) ----
    pair_scores<<<dim3(64, 8), 256, 0, stream>>>(
        spans_bf, a_vec, b_vec, w1c_t, w2_t, pw_b1, pw_b2, pw_b3, pw_w3, ts);

    // ---- adaptive S_c ----
    adaptive_sc<<<64, 32, 0, stream>>>(ts, smask, S_c);

    // ---- final loss ----
    final_loss<<<1, 32, 0, stream>>>(S_g, S_c, (float*)d_out);
}
